// Encoder_Stacked_BiLSTM_Combined_20830591385707
// MI455X (gfx1250) — matrix operand order repaired and checked by reading, not compiled, not run on hardware
//
#include <hip/hip_runtime.h>
#include <math.h>

// ---------------------------------------------------------------------------
// Problem constants (from reference): B=32, C=128, T=512, H=128, EMB=128
// ---------------------------------------------------------------------------
#define BATCH   32
#define CIN     128
#define TSTEPS  512
#define HID     128
#define GDIM    512      // 4*H gates
#define MROWS   16384    // T*B
#define HPAD    136      // HID + 8 halves pad -> conflict-free ds_load_b128

typedef __attribute__((ext_vector_type(16))) _Float16 v16h;
typedef __attribute__((ext_vector_type(8)))  _Float16 v8h;
typedef __attribute__((ext_vector_type(8)))  float    v8f;

// D = A(16x32 f16) x B(32x16 f16) + C(16x16 f32)
__device__ __forceinline__ v8f wmma_f16(v16h a, v16h b, v8f c) {
  return __builtin_amdgcn_wmma_f32_16x16x32_f16(false, a, false, b,
                                                (short)0, c, false, false);
}

// ---------------------------------------------------------------------------
// Fragment loaders.
// A (16x32, f16): lanes 0-15 hold M=0..15 with K = k0+{0..15} (elems 0..15);
//                 lanes 16-31 hold K = k0+{16..31}, same order as B.
// B (32x16, f16): lanes 0-15 hold N=0..15 with K = k0+{0..15}; lanes 16-31
//                 hold K = k0+{16..31}.  Wih / Whh are stored [N=4H][K] row
//                 major, which is exactly B^T, so B-fragment rows are
//                 contiguous in memory.
// ---------------------------------------------------------------------------
__device__ __forceinline__ v16h frag_A_g(const float* A, int ldk, int row0,
                                         int k0, int M) {
  const int lane = threadIdx.x & 31;
  int row = row0 + (lane & 15);
  if (row >= M) row = M - 1;                 // clamp (stores are guarded)
  const float* p = A + (size_t)row * ldk + k0 + ((lane >> 4) << 4);
  v16h r;
#pragma unroll
  for (int i = 0; i < 8; ++i) {
    r[i]     = (_Float16)p[i];
    r[8 + i] = (_Float16)p[8 + i];
  }
  return r;
}

__device__ __forceinline__ v16h frag_B_g(const float* W, int ldk, int n0,
                                         int k0) {
  const int lane = threadIdx.x & 31;
  const float* p = W + (size_t)(n0 + (lane & 15)) * ldk + k0 + ((lane >> 4) << 4);
  v16h r;
#pragma unroll
  for (int i = 0; i < 16; ++i) r[i] = (_Float16)p[i];
  return r;
}

__device__ __forceinline__ v16h frag_A_lds(const _Float16* sH, int ldk, int m0,
                                           int k0) {
  const int lane = threadIdx.x & 31;
  const _Float16* p = sH + (m0 + (lane & 15)) * ldk + k0 + ((lane >> 4) << 4);
  v8h a = *(const v8h*)p;          // 16B aligned: ldk=136, k0 mult of 32
  v8h b = *(const v8h*)(p + 8);
  v16h r;
#pragma unroll
  for (int i = 0; i < 8; ++i) { r[i] = a[i]; r[8 + i] = b[i]; }
  return r;
}

__device__ __forceinline__ v16h frag_B_lds(const _Float16* sW, int ldk, int n0,
                                           int k0) {
  const int lane = threadIdx.x & 31;
  const _Float16* p = sW + (n0 + (lane & 15)) * ldk + k0 + ((lane >> 4) << 4);
  v8h a = *(const v8h*)p;
  v8h b = *(const v8h*)(p + 8);
  v16h r;
#pragma unroll
  for (int i = 0; i < 8; ++i) { r[i] = a[i]; r[8 + i] = b[i]; }
  return r;
}

// ---------------------------------------------------------------------------
// [B,C,T] f32  ->  time-major rows [t*B+b][C]
// ---------------------------------------------------------------------------
__global__ __launch_bounds__(256)
void transpose_x_kernel(const float* __restrict__ x, float* __restrict__ xt) {
  int idx = blockIdx.x * 256 + threadIdx.x;     // contiguous over t (fast dim)
  int t = idx & (TSTEPS - 1);
  int c = (idx >> 9) & (CIN - 1);
  int b = idx >> 16;
  xt[((size_t)t * BATCH + b) * CIN + c] = x[idx];
}

// ---------------------------------------------------------------------------
// C[M][N] = relu?( A[M][K] @ W[N][K]^T + bias[N] )   (f16 WMMA, f32 accum)
// KSTEPS is a compile-time constant (K = 32*KSTEPS) so all K loops fully
// unroll and the per-N-tile B fragments bf[] stay in VGPRs (no scratch).
// 256 threads = 8 waves; block covers 128 rows; wave w sweeps N-tiles
// nt = w, w+8, ...
// ---------------------------------------------------------------------------
template <int KSTEPS>
__global__ __launch_bounds__(256)
void wmma_gemm_bias(const float* __restrict__ A, const float* __restrict__ W,
                    const float* __restrict__ bias, float* __restrict__ C,
                    int M, int N, int relu) {
  constexpr int K = KSTEPS << 5;
  const int wave = threadIdx.x >> 5;
  const int lane = threadIdx.x & 31;
  const int hi = lane >> 4, col = lane & 15;
  const int mbase = blockIdx.x * 128;
  const int ntiles = N >> 4;

  for (int nt = wave; nt < ntiles; nt += 8) {
    v16h bf[KSTEPS];
#pragma unroll
    for (int k = 0; k < KSTEPS; ++k) bf[k] = frag_B_g(W, K, nt << 4, k << 5);
    const float bv = bias[(nt << 4) + col];

    for (int mtile = 0; mtile < 8; ++mtile) {
      const int row0 = mbase + (mtile << 4);
      if (row0 >= M) break;
      v8f acc;
#pragma unroll
      for (int r = 0; r < 8; ++r) acc[r] = bv;
#pragma unroll
      for (int k = 0; k < KSTEPS; ++k) {
        v16h af = frag_A_g(A, K, row0, k << 5, M);
        acc = wmma_f16(af, bf[k], acc);
      }
#pragma unroll
      for (int r = 0; r < 8; ++r) {
        const int row = row0 + r + hi * 8;
        if (row < M) {
          float v = acc[r];
          if (relu) v = fmaxf(v, 0.f);
          C[(size_t)row * N + (nt << 4) + col] = v;
        }
      }
    }
  }
}

// ---------------------------------------------------------------------------
// Sequential LSTM scan, one workgroup per direction (blockIdx.x = dir).
//  - Whh (512x128) persisted in LDS as f16 (128 KB of the 320 KB/WGP).
//  - h ping-pongs between two padded 32x128 f16 LDS tiles so only ONE
//    workgroup barrier per timestep is needed (512 instead of 1024).
//  - c lives in VGPRs in the WMMA accumulator layout; never touches memory.
//  - 16 waves: wave = (mt, nh); computes gates i,f,g,o for its slice via
//    2x32x4 = 256 v_wmma per timestep across the workgroup.
//  - next timestep's pre-activations prefetched (global_prefetch_b8).
// ---------------------------------------------------------------------------
__global__ __launch_bounds__(512, 1)
void lstm_scan(const float* __restrict__ pre0, const float* __restrict__ pre1,
               const float* __restrict__ Whh0, const float* __restrict__ Whh1,
               float* __restrict__ hout, int out_ld) {
  __shared__ _Float16 sW[GDIM * HID];       // 128 KB
  __shared__ _Float16 sH[2][BATCH * HPAD];  // 2 x 8.5 KB ping-pong

  const int dir = blockIdx.x;           // 0 = forward, 1 = backward
  const float* __restrict__ pre = dir ? pre1 : pre0;
  const float* __restrict__ Whh = dir ? Whh1 : Whh0;
  const int col_off = dir * HID;

  const int tid = threadIdx.x;
  const int wave = tid >> 5;
  const int lane = tid & 31;
  const int hi = lane >> 4, col = lane & 15;
  const int mt = wave & 1;              // batch tile (rows mt*16..mt*16+15)
  const int nh = wave >> 1;             // h-column tile 0..7
  const int rowb = mt << 4;

  for (int i = tid; i < GDIM * HID; i += 512) sW[i] = (_Float16)Whh[i];
  for (int i = tid; i < BATCH * HPAD; i += 512) sH[0][i] = (_Float16)0.f;
  __syncthreads();

  v8f c;
#pragma unroll
  for (int r = 0; r < 8; ++r) c[r] = 0.f;

  for (int step = 0; step < TSTEPS; ++step) {
    const int t = dir ? (TSTEPS - 1 - step) : step;
    const float* pt = pre + (size_t)t * (BATCH * GDIM);
    const _Float16* hcur = sH[step & 1];
    _Float16* hnxt = sH[(step & 1) ^ 1];

    if (step + 1 < TSTEPS) {            // prefetch next step's 64 KB pre slice
      const int tn = dir ? (t - 1) : (t + 1);
      __builtin_prefetch(pre + (size_t)tn * (BATCH * GDIM) + tid * 32, 0, 1);
    }

    // gate pre-activations first (independent of LDS h -> overlap latency)
    v8f g4[4];
#pragma unroll
    for (int gi = 0; gi < 4; ++gi) {
      const int nt = nh + (gi << 3);
#pragma unroll
      for (int r = 0; r < 8; ++r)
        g4[gi][r] = pt[(rowb + r + hi * 8) * GDIM + (nt << 4) + col];
    }

    // h fragments for this wave's batch tile (shared across all 4 gates)
    v16h af[4];
#pragma unroll
    for (int k = 0; k < 4; ++k) af[k] = frag_A_lds(hcur, HPAD, rowb, k << 5);

    // gates i,f,g,o live at N-tiles nh, nh+8, nh+16, nh+24
#pragma unroll
    for (int gi = 0; gi < 4; ++gi) {
      const int nt = nh + (gi << 3);
      v8f acc = g4[gi];
#pragma unroll
      for (int k = 0; k < 4; ++k) {
        v16h bf = frag_B_lds(sW, HID, nt << 4, k << 5);
        acc = wmma_f16(af[k], bf, acc);
      }
      g4[gi] = acc;
    }

    // gate math entirely within the wave (TRANS pipe co-executes with WMMA)
    v8f hnew;
#pragma unroll
    for (int r = 0; r < 8; ++r) {
      const float iv = 1.f / (1.f + __expf(-g4[0][r]));
      const float fv = 1.f / (1.f + __expf(-g4[1][r]));
      const float gv = tanhf(g4[2][r]);
      const float ov = 1.f / (1.f + __expf(-g4[3][r]));
      const float cn = fv * c[r] + iv * gv;
      c[r] = cn;
      hnew[r] = ov * tanhf(cn);
    }

    // write h into the *other* LDS tile (no race with this step's readers)
    const int hc = (nh << 4) + col;
    float* po = hout + (size_t)t * BATCH * out_ld + col_off + hc;
#pragma unroll
    for (int r = 0; r < 8; ++r) {
      const int b = rowb + r + hi * 8;
      hnxt[b * HPAD + hc] = (_Float16)hnew[r];
      po[(size_t)b * out_ld] = hnew[r];
    }
    __syncthreads();                    // new h visible for next step
  }
}

// ---------------------------------------------------------------------------
// Orchestration.
// Input order: x, then 6x (Wih_f, Whh_f, b_f, Wih_b, Whh_b, b_b),
// then 6x (Wih, Whh, b), then W_out, b_out  (57 tensors).
// ---------------------------------------------------------------------------
extern "C" void kernel_launch(void* const* d_in, const int* in_sizes, int n_in,
                              void* d_out, int out_size, void* d_ws,
                              size_t ws_size, hipStream_t stream) {
  (void)in_sizes; (void)n_in; (void)out_size; (void)ws_size;

  const float* x = (const float*)d_in[0];
  float* ws = (float*)d_ws;

  float* buf0  = ws;                         // [16384][256]
  float* buf1  = buf0 + (size_t)MROWS * 256; // [16384][256]
  float* pre_f = buf1 + (size_t)MROWS * 256; // [16384][512]
  float* pre_b = pre_f + (size_t)MROWS * 512;

  transpose_x_kernel<<<(BATCH * CIN * TSTEPS) / 256, 256, 0, stream>>>(x, buf0);

  float* cur = buf0;
  float* nxt = buf1;
  int inK = CIN;

  for (int l = 0; l < 6; ++l) {             // bidirectional stack
    const float* Wih_f = (const float*)d_in[1 + l * 6 + 0];
    const float* Whh_f = (const float*)d_in[1 + l * 6 + 1];
    const float* b_f   = (const float*)d_in[1 + l * 6 + 2];
    const float* Wih_b = (const float*)d_in[1 + l * 6 + 3];
    const float* Whh_b = (const float*)d_in[1 + l * 6 + 4];
    const float* b_b   = (const float*)d_in[1 + l * 6 + 5];
    if (inK == 128) {
      wmma_gemm_bias<4><<<MROWS / 128, 256, 0, stream>>>(cur, Wih_f, b_f, pre_f,
                                                         MROWS, GDIM, 0);
      wmma_gemm_bias<4><<<MROWS / 128, 256, 0, stream>>>(cur, Wih_b, b_b, pre_b,
                                                         MROWS, GDIM, 0);
    } else {
      wmma_gemm_bias<8><<<MROWS / 128, 256, 0, stream>>>(cur, Wih_f, b_f, pre_f,
                                                         MROWS, GDIM, 0);
      wmma_gemm_bias<8><<<MROWS / 128, 256, 0, stream>>>(cur, Wih_b, b_b, pre_b,
                                                         MROWS, GDIM, 0);
    }
    lstm_scan<<<2, 512, 0, stream>>>(pre_f, pre_b, Whh_f, Whh_b, nxt, 2 * HID);
    float* tmp = cur; cur = nxt; nxt = tmp;
    inK = 2 * HID;
  }

  for (int l = 0; l < 6; ++l) {             // unidirectional stack
    const float* Wih = (const float*)d_in[37 + l * 3 + 0];
    const float* Whh = (const float*)d_in[37 + l * 3 + 1];
    const float* b   = (const float*)d_in[37 + l * 3 + 2];
    if (inK == 256)
      wmma_gemm_bias<8><<<MROWS / 128, 256, 0, stream>>>(cur, Wih, b, pre_f,
                                                         MROWS, GDIM, 0);
    else
      wmma_gemm_bias<4><<<MROWS / 128, 256, 0, stream>>>(cur, Wih, b, pre_f,
                                                         MROWS, GDIM, 0);
    lstm_scan<<<1, 512, 0, stream>>>(pre_f, pre_f, Whh, Whh, nxt, HID);
    float* tmp = cur; cur = nxt; nxt = tmp;
    inK = HID;
  }

  // head: relu(h_last @ W_out^T + b_out), h_last = rows of t = T-1
  const float* W_out = (const float*)d_in[55];
  const float* b_out = (const float*)d_in[56];
  wmma_gemm_bias<4><<<1, 256, 0, stream>>>(cur + (size_t)(TSTEPS - 1) * BATCH * HID,
                                           W_out, b_out, (float*)d_out,
                                           BATCH, 128, 1);
}
